// AttentionEntryPoint_28930899706339
// MI455X (gfx1250) — matrix operand order repaired and checked by reading, not compiled, not run on hardware
//
#include <hip/hip_runtime.h>
#include <math.h>

// ---------- types ----------
typedef __bf16 bhalf;
typedef __bf16 v16bf __attribute__((ext_vector_type(16)));
typedef __bf16 bf16x8 __attribute__((ext_vector_type(8)));
typedef float  v8f   __attribute__((ext_vector_type(8)));

#define B_  16
#define C_  512
#define S_  1024   // H*W

// ---------------------------------------------------------------------------
// CDNA5 async memory->LDS DMA (ASYNCcnt path), via inline asm so it is
// portable across ROCm 7.2 / amdgpu-toolchain (builtin arity differs).
// LDS operand = low 32 bits of the generic pointer (flat aperture rule).
// ---------------------------------------------------------------------------
__device__ __forceinline__ void async_ld_b128(void* lds, const void* gaddr) {
  unsigned loff = (unsigned)(size_t)lds;
  asm volatile("global_load_async_to_lds_b128 %0, %1, off"
               :: "v"(loff), "v"(gaddr) : "memory");
}
__device__ __forceinline__ void wait_async0() {
  asm volatile("s_wait_asynccnt 0" ::: "memory");
}

// ---------------------------------------------------------------------------
// 1) BatchNorm statistics: per-channel mean / rstd over (B, S).
// ---------------------------------------------------------------------------
__global__ __launch_bounds__(256) void bn_stats_kernel(
    const float* __restrict__ x, float* __restrict__ stats) {
  const int c = blockIdx.x;
  const int t = threadIdx.x;
  float s = 0.f, s2 = 0.f;
  for (int b = 0; b < B_; ++b) {
    const float* p = x + ((size_t)b * C_ + c) * S_;
    for (int i = t; i < S_; i += 256) {
      float v = p[i];
      s += v; s2 += v * v;
    }
  }
  __shared__ float sh1[256], sh2[256];
  sh1[t] = s; sh2[t] = s2;
  __syncthreads();
  for (int off = 128; off > 0; off >>= 1) {
    if (t < off) { sh1[t] += sh1[t + off]; sh2[t] += sh2[t + off]; }
    __syncthreads();
  }
  if (t == 0) {
    const float n = (float)(B_ * S_);
    float mean = sh1[0] / n;
    float var  = sh2[0] / n - mean * mean;
    stats[c]      = mean;
    stats[C_ + c] = rsqrtf(var + 1e-5f);
  }
}

// ---------------------------------------------------------------------------
// 2) Weight transpose + f32 -> bf16:  WT[d][c] = W[c][d]   (512x512)
// ---------------------------------------------------------------------------
__global__ __launch_bounds__(256) void wtrans_kernel(
    const float* __restrict__ W, bhalf* __restrict__ WT) {
  int idx = blockIdx.x * blockDim.x + threadIdx.x;  // = d*C_ + c
  if (idx >= C_ * C_) return;
  int d = idx / C_, c = idx % C_;
  WT[idx] = (bhalf)W[c * C_ + d];
}

// ---------------------------------------------------------------------------
// 3) Normalize + layout change [B,C,S] -> [B*S, C]; bf16 (GEMM A) + f32 (resid)
// ---------------------------------------------------------------------------
__global__ __launch_bounds__(256) void normalize_kernel(
    const float* __restrict__ x, const float* __restrict__ stats,
    bhalf* __restrict__ xnb, float* __restrict__ xnf) {
  size_t idx = (size_t)blockIdx.x * blockDim.x + threadIdx.x;  // row*C_ + c
  size_t row = idx / C_;
  int c = (int)(idx % C_);
  int b = (int)(row / S_);
  int s = (int)(row % S_);
  float v = x[((size_t)b * C_ + c) * S_ + s];
  float nv = (v - stats[c]) * stats[C_ + c];
  xnf[idx] = nv;
  xnb[idx] = (bhalf)nv;
}

// ---------------------------------------------------------------------------
// 4) Per-batch transpose of v (bf16): VT[b][d][k] = V[b][k][d]
// ---------------------------------------------------------------------------
__global__ __launch_bounds__(256) void vtrans_kernel(
    const bhalf* __restrict__ V, bhalf* __restrict__ VT) {
  size_t idx = (size_t)blockIdx.x * blockDim.x + threadIdx.x;  // b*C*S + d*S + k
  int k = (int)(idx % S_);
  size_t r = idx / S_;
  int d = (int)(r % C_);
  int b = (int)(r / C_);
  VT[idx] = V[(size_t)b * S_ * C_ + (size_t)k * C_ + d];
}

// ---------------------------------------------------------------------------
// 5) Softmax2d quirk: softmax across the BATCH axis for each (q,k).
// ---------------------------------------------------------------------------
__global__ __launch_bounds__(256) void softmax_batch_kernel(
    const float* __restrict__ sc, bhalf* __restrict__ w) {
  const size_t SS = (size_t)S_ * S_;
  size_t idx = (size_t)blockIdx.x * blockDim.x + threadIdx.x;  // q*S_ + k
  float vals[B_];
  float m = -INFINITY;
#pragma unroll
  for (int b = 0; b < B_; ++b) { vals[b] = sc[(size_t)b * SS + idx]; m = fmaxf(m, vals[b]); }
  float sum = 0.f;
#pragma unroll
  for (int b = 0; b < B_; ++b) { vals[b] = __expf(vals[b] - m); sum += vals[b]; }
  float inv = 1.f / sum;
#pragma unroll
  for (int b = 0; b < B_; ++b) w[(size_t)b * SS + idx] = (bhalf)(vals[b] * inv);
}

// ---------------------------------------------------------------------------
// 6) Batched bf16 WMMA GEMM with double-buffered ASYNC tile DMA.
//    out[M,N] = alpha * A[M,K] x Bt[N,K]^T (+resid)
//    Block tile 128x128, 8 waves, wave tile 32x64 (2x4 v8f accumulators),
//    K-step 32 via v_wmma_f32_16x16x32_bf16.  All dims multiples of 128/32.
// ---------------------------------------------------------------------------
#define BT  128
#define KT  32
#define LSTR 48   // padded LDS row stride (bf16): 96B rows -> 16B-aligned frags

// Issue the async DMA fill of one (A,B) tile pair into the given LDS buffers.
// 128 rows x 32 bf16 = 512 16-byte chunks per tile; 2 per thread per tile.
__device__ __forceinline__ void issue_tile(
    const bhalf* __restrict__ Ab, const bhalf* __restrict__ Bb,
    int rowBase, int colBase, int K, int k0,
    bhalf* __restrict__ As, bhalf* __restrict__ Bs, int t) {
#pragma unroll
  for (int it = 0; it < 2; ++it) {
    int cid = t + it * 256;
    int r   = cid >> 2;
    int sg  = cid & 3;
    async_ld_b128(&As[r * LSTR + sg * 8],
                  Ab + (size_t)(rowBase + r) * K + k0 + sg * 8);
    async_ld_b128(&Bs[r * LSTR + sg * 8],
                  Bb + (size_t)(colBase + r) * K + k0 + sg * 8);
  }
}

__global__ __launch_bounds__(256) void gemm_wmma_bf16(
    const bhalf* __restrict__ A, long long sA,
    const bhalf* __restrict__ Bt, long long sB,
    float alpha,
    float* __restrict__ outF, bhalf* __restrict__ outB,
    const float* __restrict__ resid,
    long long sO, int M, int N, int K) {
  __shared__ __align__(16) bhalf As[2][BT * LSTR];   // 2 x 12KB
  __shared__ __align__(16) bhalf Bs[2][BT * LSTR];   // 2 x 12KB

  const int t    = threadIdx.x;
  const int lane = t & 31;
  const int wave = t >> 5;
  const int wm   = (wave & 3) * 32;   // wave row base in block tile
  const int wn   = (wave >> 2) * 64;  // wave col base in block tile
  const int hi16 = (lane >= 16);
  const int lr   = lane & 15;

  const long long z = blockIdx.z;
  const bhalf* Ab = A + z * sA;
  const bhalf* Bb = Bt + z * sB;
  const int rowBase = blockIdx.y * BT;
  const int colBase = blockIdx.x * BT;

  v8f acc[2][4] = {};

  const int numK = K / KT;

  // Prologue: DMA tile 0 into buffer 0.
  issue_tile(Ab, Bb, rowBase, colBase, K, 0, As[0], Bs[0], t);
  wait_async0();
  __syncthreads();

  for (int kt = 0; kt < numK; ++kt) {
    const int cur = kt & 1;

    // DMA next tile into the other buffer while we compute on this one.
    if (kt + 1 < numK)
      issue_tile(Ab, Bb, rowBase, colBase, K, (kt + 1) * KT,
                 As[cur ^ 1], Bs[cur ^ 1], t);
    if (kt + 2 < numK) {  // global_prefetch_b8 two tiles ahead
      __builtin_prefetch(Ab + (size_t)(rowBase + (t >> 1)) * K + (kt + 2) * KT, 0, 1);
      __builtin_prefetch(Bb + (size_t)(colBase + (t >> 1)) * K + (kt + 2) * KT, 0, 1);
    }

    // A fragments (ISA 16-bit 16x32 layout):
    //   lane<16 : K {0..7, 16..23};  lane>=16 : K {8..15, 24..31}
    v16bf a2[2];
#pragma unroll
    for (int i = 0; i < 2; ++i) {
      const bhalf* p = &As[cur][(wm + i * 16 + lr) * LSTR];
      int ko = hi16 ? 8 : 0;
      bf16x8 lo = *(const bf16x8*)(p + ko);
      bf16x8 hh = *(const bf16x8*)(p + ko + 16);
#pragma unroll
      for (int e = 0; e < 8; ++e) { a2[i][e] = lo[e]; a2[i][8 + e] = hh[e]; }
    }
    // B fragments (32x16, K x N): lane<16 K {0..7, 16..23}; lane>=16 K {8..15, 24..31}
    v16bf b4[4];
#pragma unroll
    for (int j = 0; j < 4; ++j) {
      const bhalf* p = &Bs[cur][(wn + j * 16 + lr) * LSTR];
      int ko = hi16 ? 8 : 0;
      bf16x8 lo = *(const bf16x8*)(p + ko);
      bf16x8 hh = *(const bf16x8*)(p + ko + 16);
#pragma unroll
      for (int e = 0; e < 8; ++e) { b4[j][e] = lo[e]; b4[j][8 + e] = hh[e]; }
    }

#pragma unroll
    for (int i = 0; i < 2; ++i)
#pragma unroll
      for (int j = 0; j < 4; ++j)
        acc[i][j] = __builtin_amdgcn_wmma_f32_16x16x32_bf16(
            false, a2[i], false, b4[j], (short)0, acc[i][j], false, false);

    // My async DMAs of the next buffer must land before anyone reads it;
    // my reads of the current buffer were consumed by the WMMAs above.
    wait_async0();
    __syncthreads();
  }

  // Epilogue.  C/D layout: VGPR r -> M = r (+8 if lane>=16), N = lane%16.
  float* outFb      = outF  ? outF  + z * sO : nullptr;
  bhalf* outBb      = outB  ? outB  + z * sO : nullptr;
  const float* resb = resid ? resid + z * sO : nullptr;
  const int m0 = rowBase + wm + (hi16 ? 8 : 0);
  const int n0 = colBase + wn + lr;
#pragma unroll
  for (int i = 0; i < 2; ++i) {
#pragma unroll
    for (int j = 0; j < 4; ++j) {
#pragma unroll
      for (int r = 0; r < 8; ++r) {
        int m = m0 + i * 16 + r;
        int n = n0 + j * 16;
        float v = acc[i][j][r] * alpha;
        size_t o = (size_t)m * N + n;
        if (outFb) {
          if (resb) v += resb[o];
          outFb[o] = v;
        } else {
          outBb[o] = (bhalf)v;
        }
      }
    }
  }
}

// ---------------------------------------------------------------------------
// Launcher
// ---------------------------------------------------------------------------
extern "C" void kernel_launch(void* const* d_in, const int* in_sizes, int n_in,
                              void* d_out, int out_size, void* d_ws, size_t ws_size,
                              hipStream_t stream) {
  const float* x  = (const float*)d_in[0];
  const float* Qp = (const float*)d_in[1];
  const float* Kp = (const float*)d_in[2];
  const float* Vp = (const float*)d_in[3];
  float* out = (float*)d_out;

  // ---- workspace carve-up (≈220 MB) ----
  char* ws = (char*)d_ws;
  size_t off = 0;
  auto alloc = [&](size_t bytes) -> void* {
    void* p = ws + off;
    off = (off + bytes + 255) & ~(size_t)255;
    return p;
  };
  const size_t MSC = (size_t)B_ * S_ * C_;   // 8.4M elems
  const size_t SS  = (size_t)S_ * S_;        // 1M elems / batch
  float* stats = (float*)alloc(2 * C_ * sizeof(float));
  bhalf* QT    = (bhalf*)alloc((size_t)C_ * C_ * sizeof(bhalf));
  bhalf* KT_w  = (bhalf*)alloc((size_t)C_ * C_ * sizeof(bhalf));
  bhalf* VT_w  = (bhalf*)alloc((size_t)C_ * C_ * sizeof(bhalf));
  bhalf* xnb   = (bhalf*)alloc(MSC * sizeof(bhalf));
  float* xnf   = (float*)alloc(MSC * sizeof(float));
  bhalf* qb    = (bhalf*)alloc(MSC * sizeof(bhalf));
  bhalf* kb    = (bhalf*)alloc(MSC * sizeof(bhalf));
  bhalf* vb    = (bhalf*)alloc(MSC * sizeof(bhalf));
  float* sc    = (float*)alloc((size_t)B_ * SS * sizeof(float));
  bhalf* wat   = (bhalf*)alloc((size_t)B_ * SS * sizeof(bhalf));
  bhalf* vT    = (bhalf*)alloc(MSC * sizeof(bhalf));
  (void)ws_size; (void)in_sizes; (void)n_in; (void)out_size;

  // 1) batch-norm statistics
  bn_stats_kernel<<<C_, 256, 0, stream>>>(x, stats);

  // 2) weight transpose -> bf16
  const int wblk = (C_ * C_ + 255) / 256;
  wtrans_kernel<<<wblk, 256, 0, stream>>>(Qp, QT);
  wtrans_kernel<<<wblk, 256, 0, stream>>>(Kp, KT_w);
  wtrans_kernel<<<wblk, 256, 0, stream>>>(Vp, VT_w);

  // 3) normalize + layout change
  normalize_kernel<<<(int)(MSC / 256), 256, 0, stream>>>(x, stats, xnb, xnf);

  // 4) projections: [16384,512] x [512,512] -> bf16
  dim3 gp(C_ / BT, (B_ * S_) / BT, 1);
  gemm_wmma_bf16<<<gp, 256, 0, stream>>>(xnb, 0, QT,   0, 1.f, nullptr, qb, nullptr, 0, B_ * S_, C_, C_);
  gemm_wmma_bf16<<<gp, 256, 0, stream>>>(xnb, 0, KT_w, 0, 1.f, nullptr, kb, nullptr, 0, B_ * S_, C_, C_);
  gemm_wmma_bf16<<<gp, 256, 0, stream>>>(xnb, 0, VT_w, 0, 1.f, nullptr, vb, nullptr, 0, B_ * S_, C_, C_);

  // 5) scores = q k^T / sqrt(C), per batch.  k in [S,C] is already Bt[N,K].
  const float alpha = 0.044194173824159216f;  // 1/sqrt(512)
  dim3 gs(S_ / BT, S_ / BT, B_);
  gemm_wmma_bf16<<<gs, 256, 0, stream>>>(qb, (long long)S_ * C_, kb, (long long)S_ * C_,
                                         alpha, sc, nullptr, nullptr,
                                         (long long)SS, S_, S_, C_);

  // 6) softmax over the batch axis (Softmax2d quirk)
  softmax_batch_kernel<<<(int)(SS / 256), 256, 0, stream>>>(sc, wat);

  // 7) transpose v per batch for the final GEMM's Bt operand
  vtrans_kernel<<<(int)(MSC / 256), 256, 0, stream>>>(vb, vT);

  // 8) attn = w v  (+ xn residual), f32 output
  dim3 ga(C_ / BT, S_ / BT, B_);
  gemm_wmma_bf16<<<ga, 256, 0, stream>>>(wat, (long long)SS, vT, (long long)C_ * S_,
                                         1.f, out, nullptr, xnf,
                                         (long long)S_ * C_, S_, C_, S_);
}
